// PointFPSBlock_12506944766647
// MI455X (gfx1250) — compile-verified, matrix operand order checked by reading, not run on hardware
//
#include <hip/hip_runtime.h>

typedef __attribute__((ext_vector_type(16))) __bf16 v16bf;
typedef __attribute__((ext_vector_type(8)))  float  v8f;

#define WAVES  8
#define SCOLS  4
#define NPIX   128         // 32 K-positions * SCOLS columns
#define NT     8           // NPIX / 16  (N-tiles per workgroup)

// LDS row strides in elements; chosen so (stride*2/4) % 64 == 4 -> conflict-free
// b128 fragment reads across each half-wave.
#define STRIDE_X 648       // 640 + 8
#define STRIDE_0 520       // 512 + 8
#define STRIDE_1 264       // 256 + 8
#define STRIDE_2 136       // 128 + 8
#define STRIDE_3 72        //  64 + 8

// ---------------- fused GEMM layer: out[n][cout] = relu(scale*(W @ in) + bias) ----
template<int CIN, int COUT, int INSTRIDE, int OUTSTRIDE>
__device__ __forceinline__ void layer(const __bf16* __restrict__ Wb,
                                      const float*  __restrict__ scaleArr,
                                      const float*  __restrict__ biasArr,
                                      const __bf16* __restrict__ inLds,
                                      __bf16*       __restrict__ outLds,
                                      int wave, int lane)
{
  const int mrow  = lane & 15;          // row within 16 (A: M, B: N)
  const int khalf = (lane >> 4) << 3;   // +8 K-offset for lanes 16..31

  for (int mt = wave; mt < COUT / 16; mt += WAVES) {
    v8f acc[NT] = {};
    const __bf16* wp = Wb + (long)(mt * 16 + mrow) * CIN + khalf;
    for (int kb = 0; kb < CIN / 32; ++kb) {
      // A fragment: 16x32 bf16 weights, lane layout per CDNA5 ISA 7.12.2
      v16bf a;
      #pragma unroll
      for (int i = 0; i < 8; ++i) a[i] = wp[i];
      #pragma unroll
      for (int i = 0; i < 8; ++i) a[8 + i] = wp[16 + i];
      wp += 32;
      #pragma unroll
      for (int nt = 0; nt < NT; ++nt) {
        // B fragment from LDS ([n][c] layout): two contiguous 16B reads/lane
        const __bf16* bp = inLds + (nt * 16 + mrow) * INSTRIDE + kb * 32 + khalf;
        v16bf b;
        #pragma unroll
        for (int i = 0; i < 8; ++i) b[i] = bp[i];
        #pragma unroll
        for (int i = 0; i < 8; ++i) b[8 + i] = bp[16 + i];
        acc[nt] = __builtin_amdgcn_wmma_f32_16x16x32_bf16(
            false, a, false, b, (short)0, acc[nt], false, false);
      }
    }
    // Epilogue: BN (folded scale/shift) + ReLU, store bf16 -> out LDS [n][cout]
    const int mbase = mt * 16 + khalf;  // channel of accumulator register r is mbase+r
    float sc[8], bi[8];
    #pragma unroll
    for (int r = 0; r < 8; ++r) { sc[r] = scaleArr[mbase + r]; bi[r] = biasArr[mbase + r]; }
    #pragma unroll
    for (int nt = 0; nt < NT; ++nt) {
      const int n = nt * 16 + (lane & 15);
      unsigned* op = (unsigned*)(outLds + n * OUTSTRIDE + mbase);
      #pragma unroll
      for (int r = 0; r < 8; r += 2) {
        float y0 = fmaxf(fmaf(acc[nt][r],     sc[r],     bi[r]),     0.f);
        float y1 = fmaxf(fmaf(acc[nt][r + 1], sc[r + 1], bi[r + 1]), 0.f);
        __bf16 h0 = (__bf16)y0, h1 = (__bf16)y1;
        unsigned u = ((unsigned)__builtin_bit_cast(unsigned short, h1) << 16) |
                      (unsigned)__builtin_bit_cast(unsigned short, h0);
        op[r >> 1] = u;
      }
    }
  }
}

// ---------------- main fused kernel ------------------------------------------------
__global__ __launch_bounds__(256) void fps_refine_fused(
    const float* __restrict__ xyz,   // [8,3,32,1024]
    const float* __restrict__ pts,   // [8,640,32,1024]
    const __bf16* __restrict__ W0b, const __bf16* __restrict__ W1b,
    const __bf16* __restrict__ W2b, const __bf16* __restrict__ W3b,
    const float* __restrict__ sb,    // packed per-channel scale/bias
    float* __restrict__ out)         // [8,3,64,1024]
{
  __shared__ __bf16 bufA[NPIX * STRIDE_X];   // X, then layer1 out, then layer3 out
  __shared__ __bf16 bufB[NPIX * STRIDE_0];   // layer0 out, then layer2 out
  __shared__ float  xyzT[3 * NPIX];

  const int t    = threadIdx.x;
  const int wave = t >> 5;
  const int lane = t & 31;
  const int b    = blockIdx.x / (1024 / SCOLS);
  const int s0   = (blockIdx.x % (1024 / SCOLS)) * SCOLS;

  // ---- stage X (f32 -> bf16) into LDS as [n][c], n = k*SCOLS + sc ----
  {
    const float* base = pts + (long)b * 640 * 32 * 1024 + s0;
    for (int idx = t; idx < 640 * 32; idx += 256) {
      const int c = idx >> 5, k = idx & 31;
      const float4 v4 = *(const float4*)(base + ((long)c * 32 + k) * 1024);
      bufA[(k * SCOLS + 0) * STRIDE_X + c] = (__bf16)v4.x;
      bufA[(k * SCOLS + 1) * STRIDE_X + c] = (__bf16)v4.y;
      bufA[(k * SCOLS + 2) * STRIDE_X + c] = (__bf16)v4.z;
      bufA[(k * SCOLS + 3) * STRIDE_X + c] = (__bf16)v4.w;
    }
    if (t < 96) {                       // 3 channels * 32 k
      const int c = t / 32, k = t % 32;
      const float4 v4 = *(const float4*)(xyz + (((long)b * 3 + c) * 32 + k) * 1024 + s0);
      xyzT[c * NPIX + k * SCOLS + 0] = v4.x;
      xyzT[c * NPIX + k * SCOLS + 1] = v4.y;
      xyzT[c * NPIX + k * SCOLS + 2] = v4.z;
      xyzT[c * NPIX + k * SCOLS + 3] = v4.w;
    }
  }
  __syncthreads();

  const float* sc0 = sb;        const float* bi0 = sb + 512;
  const float* sc1 = sb + 1024; const float* bi1 = sb + 1280;
  const float* sc2 = sb + 1536; const float* bi2 = sb + 1664;
  const float* sc3 = sb + 1792; const float* bi3 = sb + 1856;

  layer<640, 512, STRIDE_X, STRIDE_0>(W0b, sc0, bi0, bufA, bufB, wave, lane);
  __syncthreads();
  layer<512, 256, STRIDE_0, STRIDE_1>(W1b, sc1, bi1, bufB, bufA, wave, lane);
  __syncthreads();
  layer<256, 128, STRIDE_1, STRIDE_2>(W2b, sc2, bi2, bufA, bufB, wave, lane);
  __syncthreads();
  layer<128,  64, STRIDE_2, STRIDE_3>(W3b, sc3, bi3, bufB, bufA, wave, lane);
  __syncthreads();

  // ---- softmax over K (32) + weighted sum with xyz, one thread per (d, sc) ----
  {
    const int d = t & 63, sc = t >> 6;   // 256 threads = 64 d * 4 sc
    float a[32];
    float mx = -1e30f;
    #pragma unroll
    for (int k = 0; k < 32; ++k) {
      a[k] = (float)bufA[(k * SCOLS + sc) * STRIDE_3 + d];
      mx = fmaxf(mx, a[k]);
    }
    float sum = 0.f;
    #pragma unroll
    for (int k = 0; k < 32; ++k) { a[k] = __expf(a[k] - mx); sum += a[k]; }
    const float inv = 1.f / sum;
    #pragma unroll
    for (int c = 0; c < 3; ++c) {
      float o = 0.f;
      #pragma unroll
      for (int k = 0; k < 32; ++k) o += a[k] * xyzT[c * NPIX + k * SCOLS + sc];
      out[(((long)b * 3 + c) * 64 + d) * 1024 + s0 + sc] = o * inv;
    }
  }
}

// ---------------- prep kernels -----------------------------------------------------
__global__ void prep_bn(const float* __restrict__ b, const float* __restrict__ g,
                        const float* __restrict__ be, const float* __restrict__ m,
                        const float* __restrict__ v,
                        float* __restrict__ scale, float* __restrict__ bias, int n)
{
  int i = blockIdx.x * blockDim.x + threadIdx.x;
  if (i < n) {
    float inv = g[i] * rsqrtf(v[i] + 1e-5f);
    scale[i] = inv;
    bias[i]  = b[i] * inv + be[i] - m[i] * inv;
  }
}

__global__ void cvt_bf16(const float* __restrict__ w, __bf16* __restrict__ o, int n)
{
  int i = blockIdx.x * blockDim.x + threadIdx.x;
  if (i < n) o[i] = (__bf16)w[i];
}

// ---------------- launcher ---------------------------------------------------------
extern "C" void kernel_launch(void* const* d_in, const int* in_sizes, int n_in,
                              void* d_out, int out_size, void* d_ws, size_t ws_size,
                              hipStream_t stream)
{
  (void)in_sizes; (void)n_in; (void)out_size; (void)ws_size;
  const float* xyz = (const float*)d_in[0];
  const float* pts = (const float*)d_in[1];
  const float* W[4]    = {(const float*)d_in[2],  (const float*)d_in[8],
                          (const float*)d_in[14], (const float*)d_in[20]};
  const float* bn[4][5];
  for (int l = 0; l < 4; ++l)
    for (int j = 0; j < 5; ++j)
      bn[l][j] = (const float*)d_in[3 + 6 * l + j];   // b, g, beta, m, v

  // workspace layout: bf16 weights then packed scale/bias floats
  __bf16* W0b = (__bf16*)d_ws;           // 512*640
  __bf16* W1b = W0b + 512 * 640;         // 256*512
  __bf16* W2b = W1b + 256 * 512;         // 128*256
  __bf16* W3b = W2b + 128 * 256;         //  64*128
  float*  sb  = (float*)(W3b + 64 * 128);

  const int wn[4] = {512 * 640, 256 * 512, 128 * 256, 64 * 128};
  __bf16* wb[4] = {W0b, W1b, W2b, W3b};
  for (int l = 0; l < 4; ++l)
    cvt_bf16<<<(wn[l] + 255) / 256, 256, 0, stream>>>(W[l], wb[l], wn[l]);

  const int ch[4]  = {512, 256, 128, 64};
  const int off[4] = {0, 1024, 1536, 1792};
  for (int l = 0; l < 4; ++l)
    prep_bn<<<(ch[l] + 255) / 256, 256, 0, stream>>>(
        bn[l][0], bn[l][1], bn[l][2], bn[l][3], bn[l][4],
        sb + off[l], sb + off[l] + ch[l], ch[l]);

  fps_refine_fused<<<8 * (1024 / SCOLS), 256, 0, stream>>>(
      xyz, pts, W0b, W1b, W2b, W3b, sb, (float*)d_out);
}
